// MatixModel_69526930587729
// MI455X (gfx1250) — matrix operand order repaired and checked by reading, not compiled, not run on hardware
//
#include <hip/hip_runtime.h>
#include <hip/hip_fp16.h>
#include <hip/hip_bf16.h>

typedef __attribute__((ext_vector_type(16))) _Float16 v16h;
typedef __attribute__((ext_vector_type(8)))  _Float16 v8h;
typedef __attribute__((ext_vector_type(8)))  float    v8f;

#define NB  4
#define NL  160
#define ND  768
#define NE  9
#define NM  (NB*NL)        /* 640 rows */
#define NOUT_HEAD (NM*2*NE) /* 11520 */

// ---------------------------------------------------------------------------
// Prep: f32 -> f16 copy of bert_out (row-major A matrix, [640,768])
// ---------------------------------------------------------------------------
__global__ void k_cvt_bert(const float* __restrict__ src, _Float16* __restrict__ dst) {
    int idx = blockIdx.x * blockDim.x + threadIdx.x;   // exactly 640*768 threads
    dst[idx] = (_Float16)src[idx];
}

// ---------------------------------------------------------------------------
// Prep: build transposed (and optionally combined) f16 weight matrix.
// srcA/srcB are [768,768] blocks of a [1536,768] Linear weight (k-major).
// dst[h*768 + k] = srcA[k*768+h] (+ srcB[k*768+h] if add) -> B^T layout [N,K]
// ---------------------------------------------------------------------------
__global__ void k_prep_wT(_Float16* __restrict__ dst,
                          const float* __restrict__ srcA,
                          const float* __restrict__ srcB, int add) {
    int idx = blockIdx.x * blockDim.x + threadIdx.x;   // exactly 768*768 threads
    int h = idx / ND;
    int k = idx - h * ND;
    float v = srcA[k * ND + h];
    if (add) v += srcB[k * ND + h];
    dst[idx] = (_Float16)v;
}

// ---------------------------------------------------------------------------
// Prep: w_vec[h] = sum_k rel_w[k]*(W1a[k,h]+W1b[k,h])
//       c_vec[h] = ht_b1[h] + sum_k rel_b[k]*(W1a[k,h]+W1b[k,h])
// ---------------------------------------------------------------------------
__global__ void k_vec(const float* __restrict__ relw, const float* __restrict__ relb,
                      const float* __restrict__ htw1, const float* __restrict__ htb1,
                      float* __restrict__ wv, float* __restrict__ cv) {
    int h = blockIdx.x * blockDim.x + threadIdx.x;
    if (h >= ND) return;
    float s1 = 0.f, s2 = 0.f;
    for (int k = 0; k < ND; ++k) {
        float w = htw1[k * ND + h] + htw1[(k + ND) * ND + h];
        s1 += relw[k] * w;
        s2 += relb[k] * w;
    }
    wv[h] = s1;
    cv[h] = htb1[h] + s2;
}

// ---------------------------------------------------------------------------
// WMMA GEMM: C[640,768] = A[640,768] * B[768,768] (+bias, relu)
// A: f16 row-major. Bt: f16 [N,K] (i.e., B transposed). C: f32 row-major.
// One wave per 16x16 tile; K consumed 32/step via v_wmma_f32_16x16x32_f16.
// ---------------------------------------------------------------------------
__global__ void k_gemm(const _Float16* __restrict__ A, const _Float16* __restrict__ Bt,
                       float* __restrict__ C, const float* __restrict__ bias, int do_relu) {
    const int K = ND, N = ND;
    int wid  = blockIdx.x * 8 + (threadIdx.x >> 5);
    int lane = threadIdx.x & 31;
    int mt = wid / (N / 16);
    int nt = wid - mt * (N / 16);
    int m0 = mt * 16, n0 = nt * 16;
    int g = lane >> 4;       // half-wave group
    int r = lane & 15;       // row (A) / col (B) within tile

    const _Float16* arow = A  + (size_t)(m0 + r) * K;
    const _Float16* brow = Bt + (size_t)(n0 + r) * K;

    v8f c = {};
    for (int kb = 0; kb < K; kb += 32) {
        // A 16x32 f16 fragment: lanes 0-15 need K kb+{0..7,16..23}; lanes 16-31 +8
        v8h a0 = *(const v8h*)(arow + kb + g * 8);
        v8h a1 = *(const v8h*)(arow + kb + 16 + g * 8);
        // B 32x16 fragment from Bt: lanes 0-15 K kb+{0..7,16..23}; lanes 16-31 +8
        v8h b0 = *(const v8h*)(brow + kb + g * 8);
        v8h b1 = *(const v8h*)(brow + kb + 16 + g * 8);
        v16h av = __builtin_shufflevector(a0, a1, 0,1,2,3,4,5,6,7,8,9,10,11,12,13,14,15);
        v16h bv = __builtin_shufflevector(b0, b1, 0,1,2,3,4,5,6,7,8,9,10,11,12,13,14,15);
        c = __builtin_amdgcn_wmma_f32_16x16x32_f16(false, av, false, bv,
                                                   (short)0, c, false, false);
    }

    int col = n0 + r;
    float bv_ = bias ? bias[col] : 0.0f;
#pragma unroll
    for (int j = 0; j < 8; ++j) {   // D layout: VGPR j -> M = j + 8*g, N = r
        float v = c[j] + bv_;
        if (do_relu) v = fmaxf(v, 0.0f);
        C[(size_t)(m0 + j + g * 8) * N + col] = v;
    }
}

// ---------------------------------------------------------------------------
// Small head: out[row,c] = relu_hidden[row,:] @ W2[:,c] + b2[c]  (c < 18)
// ---------------------------------------------------------------------------
__global__ void k_head(const float* __restrict__ H, const float* __restrict__ W2,
                       const float* __restrict__ b2, float* __restrict__ out) {
    int idx = blockIdx.x * blockDim.x + threadIdx.x;   // exactly 640*18 threads
    int row = idx / (2 * NE);
    int c   = idx - row * (2 * NE);
    const float* h = H + (size_t)row * ND;
    float s = b2[c];
    for (int d = 0; d < ND; ++d) s += h[d] * W2[d * (2 * NE) + c];
    out[idx] = s;
}

// ---------------------------------------------------------------------------
// Pairwise head_tail: out[b,l,m,j] =
//   sum_d relu(U[b,l,d] + V[b,m,d] + rp[b,l,m]*wv[d] + cv[d]) * w2[d,j] + b2[j]
// One block per (b,l); U-row(+cv), wv, w2 cached in LDS; wave per m, lane over d.
// ---------------------------------------------------------------------------
__global__ void k_pair(const float* __restrict__ U, const float* __restrict__ V,
                       const float* __restrict__ wv, const float* __restrict__ cv,
                       const float* __restrict__ relpos, const float* __restrict__ w2,
                       const float* __restrict__ b2, float* __restrict__ out) {
    __shared__ float  sU [ND];
    __shared__ float  sWv[ND];
    __shared__ float2 sW2[ND];

    int bl  = blockIdx.x;            // b*160 + l
    int b   = bl / NL;
    int tid = threadIdx.x;
    for (int i = tid; i < ND; i += 256) {
        sU[i]  = U[(size_t)bl * ND + i] + cv[i];
        sWv[i] = wv[i];
        sW2[i] = ((const float2*)w2)[i];
    }
    __syncthreads();

    int wave = tid >> 5, lane = tid & 31;
    float bb0 = b2[0], bb1 = b2[1];

    for (int m = wave; m < NL; m += 8) {
        float rp = relpos[(size_t)bl * NL + m];
        const float* vrow = V + ((size_t)b * NL + m) * ND;
        float s0 = 0.f, s1 = 0.f;
#pragma unroll 4
        for (int d0 = 0; d0 < ND; d0 += 32) {
            int d = d0 + lane;
            float t = fmaxf(sU[d] + vrow[d] + rp * sWv[d], 0.f);
            float2 w = sW2[d];
            s0 += t * w.x;
            s1 += t * w.y;
        }
        for (int off = 16; off > 0; off >>= 1) {   // wave32 reduction
            s0 += __shfl_xor(s0, off, 32);
            s1 += __shfl_xor(s1, off, 32);
        }
        if (lane == 0) {
            size_t o = ((size_t)bl * NL + m) * 2;
            out[o]     = s0 + bb0;
            out[o + 1] = s1 + bb1;
        }
    }
}

// ---------------------------------------------------------------------------
extern "C" void kernel_launch(void* const* d_in, const int* in_sizes, int n_in,
                              void* d_out, int out_size, void* d_ws, size_t ws_size,
                              hipStream_t stream) {
    (void)in_sizes; (void)n_in; (void)out_size; (void)ws_size;

    const float* bert  = (const float*)d_in[0];
    const float* relp  = (const float*)d_in[1];
    const float* relw  = (const float*)d_in[2];
    const float* relb  = (const float*)d_in[3];
    const float* h_w1  = (const float*)d_in[4];
    const float* h_b1  = (const float*)d_in[5];
    const float* h_w2  = (const float*)d_in[6];
    const float* h_b2  = (const float*)d_in[7];
    const float* t_w1  = (const float*)d_in[8];
    const float* t_b1  = (const float*)d_in[9];
    const float* t_w2  = (const float*)d_in[10];
    const float* t_b2  = (const float*)d_in[11];
    const float* ht_w1 = (const float*)d_in[12];
    const float* ht_b1 = (const float*)d_in[13];
    const float* ht_w2 = (const float*)d_in[14];
    const float* ht_b2 = (const float*)d_in[15];
    float* outp = (float*)d_out;

    // Workspace layout (bytes)
    char* ws = (char*)d_ws;
    const size_t szBertH = (size_t)NM * ND * sizeof(_Float16);   // 983040
    const size_t szWT    = (size_t)ND * ND * sizeof(_Float16);   // 1179648
    const size_t szF32   = (size_t)NM * ND * sizeof(float);      // 1966080
    _Float16* bertH = (_Float16*)(ws);
    _Float16* WaT   = (_Float16*)(ws + szBertH);
    _Float16* WbT   = (_Float16*)(ws + szBertH + 1 * szWT);
    _Float16* WhT   = (_Float16*)(ws + szBertH + 2 * szWT);
    _Float16* WtT   = (_Float16*)(ws + szBertH + 3 * szWT);
    char* f32base   =             ws + szBertH + 4 * szWT;
    float* U  = (float*)(f32base);
    float* V  = (float*)(f32base + 1 * szF32);
    float* Hh = (float*)(f32base + 2 * szF32);
    float* Ht = (float*)(f32base + 3 * szF32);
    float* wv = (float*)(f32base + 4 * szF32);
    float* cv = (float*)(f32base + 4 * szF32 + ND * sizeof(float));

    // ---- Prep ----
    k_cvt_bert<<<(NM * ND) / 256, 256, 0, stream>>>(bert, bertH);
    k_prep_wT <<<(ND * ND) / 256, 256, 0, stream>>>(WaT, ht_w1,            ht_w1 + (size_t)ND*ND, 0);
    k_prep_wT <<<(ND * ND) / 256, 256, 0, stream>>>(WbT, ht_w1 + (size_t)ND*ND, ht_w1,            0);
    k_prep_wT <<<(ND * ND) / 256, 256, 0, stream>>>(WhT, h_w1,             h_w1 + (size_t)ND*ND,  1);
    k_prep_wT <<<(ND * ND) / 256, 256, 0, stream>>>(WtT, t_w1,             t_w1 + (size_t)ND*ND,  1);
    k_vec<<<3, 256, 0, stream>>>(relw, relb, ht_w1, ht_b1, wv, cv);

    // ---- WMMA GEMMs: 40x48 = 1920 tiles, 8 waves/block -> 240 blocks ----
    const int gemmBlocks = (NM / 16) * (ND / 16) / 8;
    k_gemm<<<gemmBlocks, 256, 0, stream>>>(bertH, WaT, U,  nullptr, 0);
    k_gemm<<<gemmBlocks, 256, 0, stream>>>(bertH, WbT, V,  nullptr, 0);
    k_gemm<<<gemmBlocks, 256, 0, stream>>>(bertH, WhT, Hh, h_b1,    1);
    k_gemm<<<gemmBlocks, 256, 0, stream>>>(bertH, WtT, Ht, t_b1,    1);

    // ---- Small heads: pred_h_type, pred_t_type ----
    k_head<<<(NM * 2 * NE) / 256, 256, 0, stream>>>(Hh, h_w2, h_b2, outp);
    k_head<<<(NM * 2 * NE) / 256, 256, 0, stream>>>(Ht, t_w2, t_b2, outp + NOUT_HEAD);

    // ---- Pairwise head_tail ----
    k_pair<<<NM, 256, 0, stream>>>(U, V, wv, cv, relp, ht_w2, ht_b2,
                                   outp + 2 * NOUT_HEAD);
}
